// SelfAttention_19241453486132
// MI455X (gfx1250) — compile-verified, matrix operand order checked by reading, not run on hardware
//
#include <hip/hip_runtime.h>
#include <math.h>

#define D_MODEL 256
#define H_HEADS 8
#define DH      32
#define N_TOK   4608
#define M_RED   2304
#define QT      16
#define NBIN    256
#define EPS     1e-5f
#define ATT_SCALE 0.17677669529663687f   // DH^-0.5

typedef __bf16 bf16;
typedef __attribute__((ext_vector_type(16))) __bf16 bf16x16;
typedef __attribute__((ext_vector_type(8)))  __bf16 bf16x8;
typedef __attribute__((ext_vector_type(8)))  float  floatx8;
typedef __attribute__((ext_vector_type(4)))  float  floatx4;

// Native f32 -> bf16 convert: lowers to v_cvt_*bf16_f32 on gfx1250 (RTNE),
// replacing a 4-VALU-op manual rounding sequence per element.
__device__ __forceinline__ bf16 f2bf(float f) { return (bf16)f; }

// Gather a 16x32 bf16 fragment (A-layout, ISA 7.12.2) from a row-major LDS
// tile: lane L holds row (L&15), K chunks [kb,kb+8) and [kb+16,kb+24), kb=8*(L>>4).
__device__ __forceinline__ bf16x16 frag16x32_lds(const bf16* base, int ld, int lane) {
  const int r  = lane & 15;
  const int kb = (lane >> 4) << 3;
  const bf16* p = base + r * ld + kb;
  bf16x8 lo = *(const bf16x8*)(p);
  bf16x8 hi = *(const bf16x8*)(p + 16);
  return __builtin_shufflevector(lo, hi, 0,1,2,3,4,5,6,7,8,9,10,11,12,13,14,15);
}

// ---------------------------------------------------------------------------
// Kernel 1: spatial reduction (two stride-2 depthwise 2x2 convs on the
// de-interleaved 96x48 map) + LayerNorm over C.  One block per reduced token.
// ---------------------------------------------------------------------------
__global__ __launch_bounds__(256) void sr_layernorm_kernel(
    const float* __restrict__ x,     // 4608 x 256
    const float* __restrict__ Wsr1, const float* __restrict__ bsr1,
    const float* __restrict__ Wsr2, const float* __restrict__ bsr2,
    const float* __restrict__ gamma, const float* __restrict__ beta,
    float* __restrict__ xr) {        // 2304 x 256
  const int t = blockIdx.x;
  const int c = threadIdx.x;
  const int s  = t & 1;
  const int p  = t >> 1;
  const int r  = p / 24;
  const int cc = p - r * 24;

  auto mapv = [&](int row, int col) -> float {
    // rows [0,48): even tokens; rows [48,96): odd tokens
    int n = (row < 48) ? (2 * (row * 48 + col)) : (2 * ((row - 48) * 48 + col) + 1);
    return x[(size_t)n * D_MODEL + c];
  };

  float v;
  if (s == 0) {
    const float* w = Wsr1 + c * 4;
    v = mapv(2*r, 2*cc) * w[0] + mapv(2*r, 2*cc+1) * w[1] +
        mapv(2*r+1, 2*cc) * w[2] + mapv(2*r+1, 2*cc+1) * w[3] + bsr1[c];
  } else {
    const float* w = Wsr2 + c * 4;
    float a = bsr2[c];
#pragma unroll
    for (int dr = 0; dr < 2; ++dr)
#pragma unroll
      for (int dc = 0; dc < 2; ++dc) {
        int ur = 2*r + dr - 1, uc = 2*cc + dc - 1;     // top/left zero pad
        float xv = (ur < 0 || uc < 0) ? 0.0f : mapv(ur, uc);
        a += xv * w[dr*2 + dc];
      }
    v = a;
  }

  __shared__ float sbuf[256];
  sbuf[c] = v;
  __syncthreads();
  for (int off = 128; off > 0; off >>= 1) { if (c < off) sbuf[c] += sbuf[c+off]; __syncthreads(); }
  const float mu = sbuf[0] * (1.0f / 256.0f);
  __syncthreads();
  const float d = v - mu;
  sbuf[c] = d * d;
  __syncthreads();
  for (int off = 128; off > 0; off >>= 1) { if (c < off) sbuf[c] += sbuf[c+off]; __syncthreads(); }
  const float var = sbuf[0] * (1.0f / 256.0f);
  xr[(size_t)t * D_MODEL + c] = d * rsqrtf(var + EPS) * gamma[c] + beta[c];
}

// ---------------------------------------------------------------------------
// Kernel 2: generic fp32-in/fp32-out GEMM + bias using bf16 WMMA.
// Block tile 128x64, K-step 32, 8 wave32 waves (16 rows each, 4 N-tiles).
// M%128==0, N%64==0, K%32==0 for all call sites.
// ---------------------------------------------------------------------------
__global__ __launch_bounds__(256) void gemm_bias_kernel(
    const float* __restrict__ A, const float* __restrict__ B,
    const float* __restrict__ bias, float* __restrict__ C,
    int M, int N, int K) {
  __shared__ bf16 As[128 * 32];   // row-major (m,k)
  __shared__ bf16 Bs[64 * 32];    // transposed (n,k)

  const int tid  = threadIdx.x;
  const int lane = tid & 31;
  const int wave = tid >> 5;
  const int m0 = blockIdx.x * 128;
  const int n0 = blockIdx.y * 64;

  floatx8 acc[4];
#pragma unroll
  for (int i = 0; i < 4; ++i) acc[i] = (floatx8)(0.0f);

  for (int k0 = 0; k0 < K; k0 += 32) {
    // Issue ALL global loads first so the load counter can drain while
    // conversions run (avoids per-vector s_wait_loadcnt 0 stalls).
    floatx4 va[4], vb2[2];
#pragma unroll
    for (int i = 0; i < 4; ++i) {
      int j   = tid + i * 256;                   // 1024 float4 of A tile
      int row = j >> 3;
      int col = (j & 7) << 2;
      va[i] = *(const floatx4*)(A + (size_t)(m0 + row) * K + k0 + col);
    }
#pragma unroll
    for (int i = 0; i < 2; ++i) {
      int j    = tid + i * 256;                  // 512 float4 of B tile
      int krow = j >> 4;
      int ncol = (j & 15) << 2;
      vb2[i] = *(const floatx4*)(B + (size_t)(k0 + krow) * N + n0 + ncol);
    }
    if (k0 + 32 < K) {   // warm L2/L0 for the next K tile
      __builtin_prefetch(A + (size_t)(m0 + (tid >> 1)) * K + k0 + 32, 0, 1);
      __builtin_prefetch(B + (size_t)(k0 + 32 + (tid & 31)) * N + n0, 0, 1);
    }
    // convert + stage A (row-major m,k)
#pragma unroll
    for (int i = 0; i < 4; ++i) {
      int j   = tid + i * 256;
      int row = j >> 3;
      int col = (j & 7) << 2;
      bf16* d = As + row * 32 + col;
      d[0] = f2bf(va[i].x); d[1] = f2bf(va[i].y);
      d[2] = f2bf(va[i].z); d[3] = f2bf(va[i].w);
    }
    // convert + stage B transposed (n,k)
#pragma unroll
    for (int i = 0; i < 2; ++i) {
      int j    = tid + i * 256;
      int krow = j >> 4;
      int ncol = (j & 15) << 2;
      Bs[(ncol+0)*32 + krow] = f2bf(vb2[i].x);
      Bs[(ncol+1)*32 + krow] = f2bf(vb2[i].y);
      Bs[(ncol+2)*32 + krow] = f2bf(vb2[i].z);
      Bs[(ncol+3)*32 + krow] = f2bf(vb2[i].w);
    }
    __syncthreads();

    bf16x16 a = frag16x32_lds(As + wave * 16 * 32, 32, lane);
#pragma unroll
    for (int nt = 0; nt < 4; ++nt) {
      bf16x16 b = frag16x32_lds(Bs + nt * 16 * 32, 32, lane);
      acc[nt] = __builtin_amdgcn_wmma_f32_16x16x32_bf16(
          false, a, false, b, (short)0, acc[nt], false, false);
    }
    __syncthreads();
  }

  // epilogue: C/D layout -> lane holds col (lane&15), rows i + 8*(lane>>4)
  const int nl = lane & 15;
  const int mo = (lane >> 4) << 3;
#pragma unroll
  for (int nt = 0; nt < 4; ++nt) {
    const int nn = n0 + nt * 16 + nl;
    const float bv = bias ? bias[nn] : 0.0f;
#pragma unroll
    for (int i = 0; i < 8; ++i) {
      const int mm = m0 + wave * 16 + mo + i;
      C[(size_t)mm * N + nn] = acc[nt][i] + bv;
    }
  }
}

// ---------------------------------------------------------------------------
// Kernel 3: fused attention for one (head, 16-query tile).
// Scores kept in dynamic LDS (16 x 2304 fp32); top-k via 256-bin histogram
// select per row; masked softmax; P·V with bf16 WMMA; cross-wave reduce.
// ---------------------------------------------------------------------------
__global__ __launch_bounds__(256) void attention_kernel(
    const float* __restrict__ q,    // 4608 x 256 (head h at cols h*32..)
    const float* __restrict__ kv,   // 2304 x 512 (k at col h*32, v at 256+h*32)
    float* __restrict__ o) {        // 4608 x 256
  extern __shared__ char smem_raw[];
  float* S      = (float*)smem_raw;                 // QT * 2304
  bf16*  Qs     = (bf16*)(S + QT * M_RED);          // QT * 32
  float* rowMax = (float*)(Qs + QT * 32);
  float* rowMin = rowMax + QT;
  float* rowThr = rowMin + QT;
  float* rowInv = rowThr + QT;
  float* red2   = rowInv + QT;                      // QT*16
  float* red3   = red2 + QT * 16;                   // QT*16
  int*   hist   = (int*)(red3 + QT * 16);           // QT*256
  float* opart  = (float*)(hist + QT * NBIN);       // 8*QT*32

  const int tid  = threadIdx.x;
  const int lane = tid & 31;
  const int wave = tid >> 5;
  const int h    = blockIdx.y;
  const int q0   = blockIdx.x * QT;
  const int hc   = h * DH;

  // stage Q tile (16x32) as bf16
  for (int i = tid; i < QT * 32; i += 256) {
    int rr = i >> 5, dd = i & 31;
    Qs[rr * 32 + dd] = f2bf(q[(size_t)(q0 + rr) * D_MODEL + hc + dd]);
  }
  __syncthreads();

  const bf16x16 afrag = frag16x32_lds(Qs, 32, lane);

  // Phase 1: S = (Q K^T) * scale ; each wave owns 288 contiguous keys.
  const int keysPerWave = M_RED / 8;       // 288
  const int kbase = wave * keysPerWave;
  for (int j = 0; j < keysPerWave; j += 16) {
    const int n0k = kbase + j;
    const int keyn = n0k + (lane & 15);
    const int kb   = (lane >> 4) << 3;
    const float* kp = kv + (size_t)keyn * (2 * D_MODEL) + hc + kb;
    floatx4 v0 = *(const floatx4*)(kp);
    floatx4 v1 = *(const floatx4*)(kp + 4);
    floatx4 v2 = *(const floatx4*)(kp + 16);
    floatx4 v3 = *(const floatx4*)(kp + 20);
    bf16x16 bfrag;
    bfrag[0]=f2bf(v0.x);  bfrag[1]=f2bf(v0.y);  bfrag[2]=f2bf(v0.z);  bfrag[3]=f2bf(v0.w);
    bfrag[4]=f2bf(v1.x);  bfrag[5]=f2bf(v1.y);  bfrag[6]=f2bf(v1.z);  bfrag[7]=f2bf(v1.w);
    bfrag[8]=f2bf(v2.x);  bfrag[9]=f2bf(v2.y);  bfrag[10]=f2bf(v2.z); bfrag[11]=f2bf(v2.w);
    bfrag[12]=f2bf(v3.x); bfrag[13]=f2bf(v3.y); bfrag[14]=f2bf(v3.z); bfrag[15]=f2bf(v3.w);

    floatx8 acc = (floatx8)(0.0f);
    acc = __builtin_amdgcn_wmma_f32_16x16x32_bf16(
        false, afrag, false, bfrag, (short)0, acc, false, false);

    const int nl = lane & 15, mo = (lane >> 4) << 3;
#pragma unroll
    for (int i2 = 0; i2 < 8; ++i2)
      S[(mo + i2) * M_RED + n0k + nl] = acc[i2] * ATT_SCALE;
  }
  __syncthreads();

  // Phase 2: per-row max/min (16 threads per row)
  {
    const int rr = tid >> 4, cl = tid & 15;
    float mx = -__builtin_inff(), mn = __builtin_inff();
    for (int jj = cl; jj < M_RED; jj += 16) {
      float sv = S[rr * M_RED + jj];
      mx = fmaxf(mx, sv); mn = fminf(mn, sv);
    }
    red2[rr * 16 + cl] = mx; red3[rr * 16 + cl] = mn;
  }
  for (int i = tid; i < QT * NBIN; i += 256) hist[i] = 0;
  __syncthreads();
  if (tid < QT) {
    float mx = -__builtin_inff(), mn = __builtin_inff();
    for (int jj = 0; jj < 16; ++jj) {
      mx = fmaxf(mx, red2[tid * 16 + jj]);
      mn = fminf(mn, red3[tid * 16 + jj]);
    }
    rowMax[tid] = mx; rowMin[tid] = mn;
  }
  __syncthreads();

  // Phase 3: histogram select for the top-k (k = 0.8*M) threshold
  {
    const int rr = tid >> 4, cl = tid & 15;
    const float mn = rowMin[rr];
    const float sc = 255.0f / fmaxf(rowMax[rr] - mn, 1e-20f);
    for (int jj = cl; jj < M_RED; jj += 16) {
      int b = (int)((S[rr * M_RED + jj] - mn) * sc);
      b = min(max(b, 0), NBIN - 1);
      atomicAdd(&hist[rr * NBIN + b], 1);
    }
  }
  __syncthreads();
  if (tid < QT) {
    const int topk = (int)(M_RED * 0.8f);        // 1843
    int cnt = 0, b = NBIN - 1;
    for (; b >= 0; --b) { cnt += hist[tid * NBIN + b]; if (cnt >= topk) break; }
    const float mn = rowMin[tid];
    const float binw = fmaxf(rowMax[tid] - mn, 1e-20f) / 255.0f;
    rowThr[tid] = mn + (float)b * binw;
  }
  __syncthreads();

  // Phase 4: masked exp + row sums (P left unnormalized in S)
  {
    const int rr = tid >> 4, cl = tid & 15;
    const float mx = rowMax[rr], thr = rowThr[rr];
    float ps = 0.0f;
    for (int jj = cl; jj < M_RED; jj += 16) {
      const float sv = S[rr * M_RED + jj];
      const float pv = (sv >= thr) ? __expf(sv - mx) : 0.0f;
      S[rr * M_RED + jj] = pv;
      ps += pv;
    }
    red2[rr * 16 + cl] = ps;
  }
  __syncthreads();
  if (tid < QT) {
    float ssum = 0.0f;
    for (int jj = 0; jj < 16; ++jj) ssum += red2[tid * 16 + jj];
    rowInv[tid] = 1.0f / ssum;
  }
  __syncthreads();

  // Phase 5: O = P·V (contraction over this wave's 288 keys, chunks of 32)
  floatx8 oacc0 = (floatx8)(0.0f), oacc1 = (floatx8)(0.0f);
  for (int j = 0; j < keysPerWave; j += 32) {
    const int kk0 = kbase + j;
    bf16x16 pa;
    {
      const int rr2 = lane & 15;
      const int kb2 = (lane >> 4) << 3;
      const float* sp = S + rr2 * M_RED + kk0 + kb2;
#pragma unroll
      for (int ii = 0; ii < 8; ++ii) { pa[ii] = f2bf(sp[ii]); pa[8+ii] = f2bf(sp[16+ii]); }
    }
#pragma unroll
    for (int half = 0; half < 2; ++half) {
      const int dcol = half * 16 + (lane & 15);
      const int kb2  = (lane >> 4) << 3;
      bf16x16 vb;
#pragma unroll
      for (int ii = 0; ii < 8; ++ii) {
        vb[ii]     = f2bf(kv[(size_t)(kk0 + kb2 + ii)      * (2*D_MODEL) + D_MODEL + hc + dcol]);
        vb[8 + ii] = f2bf(kv[(size_t)(kk0 + kb2 + 16 + ii) * (2*D_MODEL) + D_MODEL + hc + dcol]);
      }
      if (half == 0)
        oacc0 = __builtin_amdgcn_wmma_f32_16x16x32_bf16(false, pa, false, vb, (short)0, oacc0, false, false);
      else
        oacc1 = __builtin_amdgcn_wmma_f32_16x16x32_bf16(false, pa, false, vb, (short)0, oacc1, false, false);
    }
  }
  // cross-wave reduction of partial O tiles
  {
    const int nl = lane & 15, mo = (lane >> 4) << 3;
#pragma unroll
    for (int i2 = 0; i2 < 8; ++i2) {
      opart[(wave * QT + mo + i2) * 32 + nl]      = oacc0[i2];
      opart[(wave * QT + mo + i2) * 32 + 16 + nl] = oacc1[i2];
    }
  }
  __syncthreads();
  for (int i = tid; i < QT * 32; i += 256) {
    const int rr = i >> 5, dd = i & 31;
    float sum = 0.0f;
#pragma unroll
    for (int w2 = 0; w2 < 8; ++w2) sum += opart[(w2 * QT + rr) * 32 + dd];
    o[(size_t)(q0 + rr) * D_MODEL + hc + dd] = sum * rowInv[rr];
  }
}

// ---------------------------------------------------------------------------
extern "C" void kernel_launch(void* const* d_in, const int* in_sizes, int n_in,
                              void* d_out, int out_size, void* d_ws, size_t ws_size,
                              hipStream_t stream) {
  const float* x    = (const float*)d_in[0];
  const float* Wq   = (const float*)d_in[1];
  const float* bq   = (const float*)d_in[2];
  const float* Wkv  = (const float*)d_in[3];
  const float* bkv  = (const float*)d_in[4];
  const float* Wp   = (const float*)d_in[5];
  const float* bp   = (const float*)d_in[6];
  const float* Wsr1 = (const float*)d_in[7];
  const float* bsr1 = (const float*)d_in[8];
  const float* Wsr2 = (const float*)d_in[9];
  const float* bsr2 = (const float*)d_in[10];
  const float* gamma= (const float*)d_in[11];
  const float* beta = (const float*)d_in[12];

  float* xr  = (float*)d_ws;                               // 2304*256
  float* qb  = xr  + (size_t)M_RED * D_MODEL;              // 4608*256
  float* kvb = qb  + (size_t)N_TOK * D_MODEL;              // 2304*512
  float* ob  = kvb + (size_t)M_RED * 2 * D_MODEL;          // 4608*256

  sr_layernorm_kernel<<<M_RED, 256, 0, stream>>>(x, Wsr1, bsr1, Wsr2, bsr2, gamma, beta, xr);

  gemm_bias_kernel<<<dim3(N_TOK / 128, D_MODEL / 64), 256, 0, stream>>>(
      x, Wq, bq, qb, N_TOK, D_MODEL, D_MODEL);
  gemm_bias_kernel<<<dim3(M_RED / 128, (2 * D_MODEL) / 64), 256, 0, stream>>>(
      xr, Wkv, bkv, kvb, M_RED, 2 * D_MODEL, D_MODEL);

  const size_t smem =
      (size_t)QT * M_RED * 4 +        // S
      (size_t)QT * 32 * 2 +           // Qs
      4 * QT * 4 +                    // rowMax/Min/Thr/Inv
      2 * QT * 16 * 4 +               // red2/red3
      (size_t)QT * NBIN * 4 +         // hist
      (size_t)8 * QT * 32 * 4;        // opart   => 183552 B (< 320KB WGP LDS)
  (void)hipFuncSetAttribute((const void*)attention_kernel,
                            hipFuncAttributeMaxDynamicSharedMemorySize, (int)smem);
  attention_kernel<<<dim3(N_TOK / QT, H_HEADS), 256, smem, stream>>>(qb, kvb, ob);

  gemm_bias_kernel<<<dim3(N_TOK / 128, D_MODEL / 64), 256, 0, stream>>>(
      ob, Wp, bp, (float*)d_out, N_TOK, D_MODEL, D_MODEL);
}
